// TemplateLayer_87101936763399
// MI455X (gfx1250) — matrix operand order repaired and checked by reading, not compiled, not run on hardware
//
#include <hip/hip_runtime.h>

#define NFACES 40000
#define NEDGES 60000
#define NNZC   120000
#define CH     512

typedef __attribute__((ext_vector_type(16))) __bf16 v16bf;
typedef __attribute__((ext_vector_type(8)))  float  v8f;
typedef __attribute__((ext_vector_type(4)))  int    v4i;

#define AS1 __attribute__((address_space(1)))
#define AS3 __attribute__((address_space(3)))

struct BfFrag { uint4 lo, hi; };

static __device__ __forceinline__ v16bf frag_cast(uint4 lo, uint4 hi) {
  BfFrag t{lo, hi};
  return __builtin_bit_cast(v16bf, t);
}

// fp32 -> bf16 via native __bf16 (backend picks the hw cvt op, RNE)
static __device__ __forceinline__ unsigned short f2bf(float f) {
  __bf16 b = (__bf16)f;
  return __builtin_bit_cast(unsigned short, b);
}

// ---- CDNA5 async global->LDS (ASYNCcnt-tracked), 16B granule ----
static __device__ __forceinline__ void async_b128(const void* g, void* l) {
#if __has_builtin(__builtin_amdgcn_global_load_async_to_lds_b128)
  __builtin_amdgcn_global_load_async_to_lds_b128(
      (AS1 v4i*)(unsigned long long)g,
      (AS3 v4i*)(unsigned)(unsigned long long)l, 0, 0);
#else
  unsigned lds_off = (unsigned)(unsigned long long)l;
  asm volatile("global_load_async_to_lds_b128 %0, %1, off"
               :: "v"(lds_off), "v"((unsigned long long)g) : "memory");
#endif
}

template<short N>
static __device__ __forceinline__ void wait_async() {
#if __has_builtin(__builtin_amdgcn_s_wait_asynccnt)
  __builtin_amdgcn_s_wait_asynccnt(N);
#else
  asm volatile("s_wait_asynccnt %0" :: "i"(N) : "memory");
#endif
}

// ---------------------------------------------------------------------------
// Tiled bf16 WMMA GEMM:  C[M,512] = A[M,512] * Bt^T   (Bt = W pre-transposed,
// K contiguous). Block tile 128x128, 8 waves, wave tile 32x64 (2x4 of 16x16),
// K-step 32, double-buffered LDS fed by async-to-LDS loads.
// SCATTER epilogue: faces->edges atomic scatter (3 contiguous nnz per row).
// ---------------------------------------------------------------------------
template<bool A_BF16, bool SCATTER>
__global__ __launch_bounds__(256) void gemm_bf16_wmma(
    const void* __restrict__ Aany, const unsigned short* __restrict__ Bt,
    float* __restrict__ C, int M,
    const int* __restrict__ eidx, const float* __restrict__ vals,
    float* __restrict__ Hacc)
{
  __shared__ unsigned short lA[2][128 * 40];   // 128 rows x 32 K, pad->40
  __shared__ unsigned short lB[2][128 * 40];   // 128 cols x 32 K, pad->40

  const int tid   = threadIdx.x;
  const int lane  = tid & 31;
  const int w     = tid >> 5;
  const int ln    = lane & 15;
  const int hs    = (lane >> 4) & 1;      // half-wave select
  const int wm0   = (w >> 1) * 32;
  const int wn0   = (w & 1) * 64;
  const int mBase = blockIdx.x * 128;
  const int nBase = blockIdx.y * 128;

  const float*          Af = (const float*)Aany;
  const unsigned short* Ab = (const unsigned short*)Aany;

  // async instructions issued per wave per stage (ASYNCcnt is per-wave)
  constexpr short PEND = A_BF16 ? 4 : 2;

  v8f acc[2][4] = {};

  auto stage = [&](int buf, int kb) {
    // B tile: always bf16, always async
    #pragma unroll
    for (int t = 0; t < 2; ++t) {
      int idx = tid + t * 256;
      int n = idx >> 2, col = (idx & 3) * 8;
      async_b128(Bt + (size_t)(nBase + n) * CH + kb + col,
                 &lB[buf][n * 40 + col]);
    }
    if constexpr (A_BF16) {
      #pragma unroll
      for (int t = 0; t < 2; ++t) {
        int idx = tid + t * 256;
        int row = idx >> 2, col = (idx & 3) * 8;
        int gm = mBase + row;
        if (gm >= M) gm = M - 1;   // clamp: those rows are discarded in epilogue
        async_b128(Ab + (size_t)gm * CH + kb + col,
                   &lA[buf][row * 40 + col]);
      }
    } else {
      // fp32 A: convert to bf16 in VALU while staging
      #pragma unroll
      for (int t = 0; t < 4; ++t) {
        int idx = tid + t * 256;
        int row = idx >> 3, col = (idx & 7) * 4;
        int gm = mBase + row;
        float4 v = make_float4(0.f, 0.f, 0.f, 0.f);
        if (gm < M) v = *(const float4*)(Af + (size_t)gm * CH + kb + col);
        unsigned int p0 = (unsigned)f2bf(v.x) | ((unsigned)f2bf(v.y) << 16);
        unsigned int p1 = (unsigned)f2bf(v.z) | ((unsigned)f2bf(v.w) << 16);
        *(uint2*)&lA[buf][row * 40 + col] = make_uint2(p0, p1);
      }
    }
  };

  auto mma_step = [&](int buf) {
    // ---- fragments per ISA 7.12.2 lane layouts ----
    const int aC0 = hs ? 8 : 0;         // A: lanes16-31 hold K 8..15 / 24..31
    const int bK0 = hs ? 8 : 0;         // B: lanes16-31 hold K 8..15 / 24..31
    v16bf afr[2], bfr[4];
    #pragma unroll
    for (int wm = 0; wm < 2; ++wm) {
      int r = (wm0 + wm * 16 + ln) * 40;
      afr[wm] = frag_cast(*(const uint4*)&lA[buf][r + aC0],
                          *(const uint4*)&lA[buf][r + 16 + aC0]);
    }
    #pragma unroll
    for (int wn = 0; wn < 4; ++wn) {
      int r = (wn0 + wn * 16 + ln) * 40;
      bfr[wn] = frag_cast(*(const uint4*)&lB[buf][r + bK0],
                          *(const uint4*)&lB[buf][r + bK0 + 16]);
    }
    #pragma unroll
    for (int wm = 0; wm < 2; ++wm)
      #pragma unroll
      for (int wn = 0; wn < 4; ++wn)
        acc[wm][wn] = __builtin_amdgcn_wmma_f32_16x16x32_bf16(
            false, afr[wm], false, bfr[wn], (short)0, acc[wm][wn], false, false);
  };

  stage(0, 0);                          // prologue: fill buffer 0

  // steady state: stage(next) overlaps with wmma(cur); last iteration peeled
  for (int kb = 0; kb < CH - 32; kb += 32) {
    const int cur = (kb >> 5) & 1;
    stage(cur ^ 1, kb + 32);            // prefetch next tile into other buffer
    wait_async<PEND>();                 // previous tile complete (newest remain)
    __syncthreads();                    // all waves have current tile in LDS
    mma_step(cur);
    __syncthreads();                    // reads of `cur` done before rewrite
  }
  wait_async<0>();                      // drain
  __syncthreads();
  mma_step(((CH - 32) >> 5) & 1);       // last tile

  // ---- epilogue ----
  #pragma unroll
  for (int wm = 0; wm < 2; ++wm) {
    #pragma unroll
    for (int wn = 0; wn < 4; ++wn) {
      int nCol = nBase + wn0 + wn * 16 + ln;
      #pragma unroll
      for (int r = 0; r < 8; ++r) {
        int m = mBase + wm0 + wm * 16 + r + (hs ? 8 : 0);
        if (m < M) {
          float v = acc[wm][wn][r];
          if constexpr (SCATTER) {
            #pragma unroll
            for (int j = 0; j < 3; ++j) {
              int nz = 3 * m + j;                       // nnz grouped by face
              atomicAdd(&Hacc[(size_t)eidx[nz] * CH + nCol], vals[nz] * v);
            }
          } else {
            C[(size_t)m * CH + nCol] = v;
          }
        }
      }
    }
  }
}

// ---------------------------------------------------------------------------
// Small helper kernels
// ---------------------------------------------------------------------------
__global__ __launch_bounds__(256) void k_zero(float* __restrict__ p, long n) {
  long i = (long)blockIdx.x * 256 + threadIdx.x;
  if (i < n) p[i] = 0.f;
}

// W [512x512] fp32 -> Wt [512x512] bf16, transposed (K contiguous per out row)
__global__ __launch_bounds__(256) void k_wt(const float* __restrict__ W,
                                            unsigned short* __restrict__ Wt) {
  int idx = blockIdx.x * 256 + threadIdx.x;
  int k = idx >> 9, n = idx & 511;
  Wt[(size_t)n * CH + k] = f2bf(W[idx]);
}

__global__ __launch_bounds__(256) void k_deg_e(const int* __restrict__ e,
                                               const float* __restrict__ v,
                                               float* __restrict__ deg, int nnz) {
  int i = blockIdx.x * 256 + threadIdx.x;
  if (i < nnz) atomicAdd(&deg[e[i]], v[i]);
}

__global__ __launch_bounds__(256) void k_deg_f(const float* __restrict__ v,
                                               float* __restrict__ degf, int nf) {
  int f = blockIdx.x * 256 + threadIdx.x;
  if (f < nf) degf[f] = v[3 * f] + v[3 * f + 1] + v[3 * f + 2];
}

// sigmoid(h/deg) and cast to bf16
__global__ __launch_bounds__(256) void k_sig_e(const float* __restrict__ hacc,
                                               const float* __restrict__ deg,
                                               unsigned short* __restrict__ hbf,
                                               long total) {
  long i = (long)blockIdx.x * 256 + threadIdx.x;
  if (i < total) {
    float d = deg[i >> 9];
    float s = 1.f / (1.f + __expf(-(hacc[i] / d)));
    hbf[i] = f2bf(s);
  }
}

// edges->faces gather (3 contiguous nnz per face) + sigmoid
__global__ __launch_bounds__(256) void k_out(const float* __restrict__ t2,
                                             const int* __restrict__ e,
                                             const float* __restrict__ v,
                                             const float* __restrict__ degf,
                                             float* __restrict__ out, long total) {
  long i = (long)blockIdx.x * 256 + threadIdx.x;
  if (i < total) {
    long f = i >> 9;
    int  c = (int)(i & 511);
    float s = 0.f;
    #pragma unroll
    for (int j = 0; j < 3; ++j) {
      long nz = 3 * f + j;
      s += v[nz] * t2[(size_t)e[nz] * CH + c];
    }
    out[i] = 1.f / (1.f + __expf(-(s / degf[f])));
  }
}

// ---------------------------------------------------------------------------
extern "C" void kernel_launch(void* const* d_in, const int* in_sizes, int n_in,
                              void* d_out, int out_size, void* d_ws, size_t ws_size,
                              hipStream_t stream) {
  const float* x    = (const float*)d_in[0];
  const float* W1   = (const float*)d_in[1];
  const float* W2   = (const float*)d_in[2];
  const int*   eidx = (const int*)d_in[3];
  /* d_in[4] = face_idx: structurally face f owns nnz 3f..3f+2 (sorted COO) */
  const float* vals = (const float*)d_in[5];
  float* out = (float*)d_out;

  char* ws = (char*)d_ws;
  size_t off = 0;
  auto alloc = [&](size_t bytes) {
    size_t o = off;
    off += (bytes + 255) & ~(size_t)255;
    return o;
  };
  unsigned short* w1t  = (unsigned short*)(ws + alloc((size_t)CH * CH * 2));
  unsigned short* w2t  = (unsigned short*)(ws + alloc((size_t)CH * CH * 2));
  float*          dege = (float*)(ws + alloc((size_t)NEDGES * 4));
  float*          degf = (float*)(ws + alloc((size_t)NFACES * 4));
  float*          hacc = (float*)(ws + alloc((size_t)NEDGES * CH * 4));
  unsigned short* hbf  = (unsigned short*)(ws + alloc((size_t)NEDGES * CH * 2));
  float*          t2   = hacc;   // hacc is dead after k_sig_e: alias as GEMM2 out

  // weights: transpose + bf16 cast
  k_wt<<<(CH * CH) / 256, 256, 0, stream>>>(W1, w1t);
  k_wt<<<(CH * CH) / 256, 256, 0, stream>>>(W2, w2t);

  // degrees
  k_zero<<<(NEDGES + 255) / 256, 256, 0, stream>>>(dege, NEDGES);
  k_deg_e<<<(NNZC + 255) / 256, 256, 0, stream>>>(eidx, vals, dege, NNZC);
  k_deg_f<<<(NFACES + 255) / 256, 256, 0, stream>>>(vals, degf, NFACES);

  // zero edge accumulator
  long htot = (long)NEDGES * CH;
  k_zero<<<(unsigned)((htot + 255) / 256), 256, 0, stream>>>(hacc, htot);

  // GEMM1: (x @ W1) fused with faces->edges scatter into hacc
  dim3 g1((NFACES + 127) / 128, CH / 128);
  gemm_bf16_wmma<false, true><<<g1, 256, 0, stream>>>(
      x, w1t, nullptr, NFACES, eidx, vals, hacc);

  // sigmoid(h/deg_e) -> bf16
  k_sig_e<<<(unsigned)((htot + 255) / 256), 256, 0, stream>>>(hacc, dege, hbf, htot);

  // GEMM2: (h @ W2) -> t2
  dim3 g2((NEDGES + 127) / 128, CH / 128);
  gemm_bf16_wmma<true, false><<<g2, 256, 0, stream>>>(
      hbf, w2t, t2, NEDGES, nullptr, nullptr, nullptr);

  // gather edges->faces + sigmoid -> out
  long otot = (long)NFACES * CH;
  k_out<<<(unsigned)((otot + 255) / 256), 256, 0, stream>>>(
      t2, eidx, vals, degf, out, otot);

  (void)in_sizes; (void)n_in; (void)out_size; (void)ws_size;
}
